// LSTMSentimentModel_10496900071463
// MI455X (gfx1250) — matrix operand order repaired and checked by reading, not compiled, not run on hardware
//
#include <hip/hip_runtime.h>
#include <hip/hip_bf16.h>

#define B_ 2048
#define L_ 512
#define T_ 256
#define E_ 32
#define C_ 32
#define H_ 32
#define G_ 128   // 4*H

typedef __attribute__((ext_vector_type(16))) __bf16   bhalf16;
typedef __attribute__((ext_vector_type(8)))  float    f32x8;
typedef __attribute__((ext_vector_type(4)))  float    f32x4;
typedef __attribute__((ext_vector_type(8)))  unsigned short u16x8;
typedef __attribute__((ext_vector_type(4)))  unsigned short u16x4;

union BFrag {
    bhalf16 v;
    u16x8 h[2];
    unsigned short s[16];
};

__device__ __forceinline__ unsigned short f2bf(float f) {
    union { float f; unsigned u; } cv; cv.f = f;
    unsigned u = cv.u;
    u += 0x7FFFu + ((u >> 16) & 1u);   // round-to-nearest-even
    return (unsigned short)(u >> 16);
}

__device__ __forceinline__ float frcp(float x) { return __builtin_amdgcn_rcpf(x); }
__device__ __forceinline__ float sigm(float x) { return frcp(1.0f + __expf(-x)); }
__device__ __forceinline__ float tanh_(float x) {
#if __has_builtin(__builtin_amdgcn_tanhf)
    return __builtin_amdgcn_tanhf(x);          // gfx1250 v_tanh_f32
#else
    float e = __expf(2.0f * x);
    return (e - 1.0f) * frcp(e + 1.0f);
#endif
}

// A-matrix 16x32 bf16 fragment: lane (m = lid&15, hi = lid>>4):
// halves 0..15 = A[m][hi*16 + 0..15] (contiguous k order, matches B fragments)
__device__ __forceinline__ bhalf16 load_a_frag(const unsigned short* A, int ld, int lid) {
    int m = lid & 15, hi = lid >> 4;
    BFrag fr;
    fr.h[0] = *(const u16x8*)(A + m * ld + hi * 16);
    fr.h[1] = *(const u16x8*)(A + m * ld + hi * 16 + 8);
    return fr.v;
}

// B-matrix 32x16 bf16 fragment where B = W^T, W row-major [N][32]:
// lane (n = lid&15, kb = (lid>>4)*16) holds B[kb+j][n] = W[n][kb+j], contiguous.
__device__ __forceinline__ bhalf16 load_b_frag_wT(const unsigned short* W, int lid) {
    int n = lid & 15, kb = (lid >> 4) * 16;
    BFrag fr;
    fr.h[0] = *(const u16x8*)(W + n * 32 + kb);
    fr.h[1] = *(const u16x8*)(W + n * 32 + kb + 8);
    return fr.v;
}

// B fragment from pre-swizzled (fragment-ordered) weights: lane's 16 halves
// are contiguous at fw + lane*16.
__device__ __forceinline__ bhalf16 load_b_frag_swz(const unsigned short* fw, int lid) {
    const unsigned short* p = fw + lid * 16;
    BFrag fr;
    fr.h[0] = *(const u16x8*)(p);
    fr.h[1] = *(const u16x8*)(p + 8);
    return fr.v;
}

__device__ __forceinline__ f32x8 wmma_bf16(bhalf16 a, bhalf16 b, f32x8 c) {
    return __builtin_amdgcn_wmma_f32_16x16x32_bf16(false, a, false, b, (short)0, c, false, false);
}

// ---------------------------------------------------------------------------
// K0: weight conversion f32 -> bf16.
// conv_w [C][E][3] -> fragment-ordered fragW[kk][chtile][lane][j]:
//   ch = chtile*16 + (lane&15), e = (lane>>4)*16 + j  (B[e][ch] = w[ch][e][kk])
// ---------------------------------------------------------------------------
__global__ void wprep_kernel(const float* __restrict__ wih_f, const float* __restrict__ whh_f,
                             const float* __restrict__ wih_b, const float* __restrict__ whh_b,
                             const float* __restrict__ conv_w,
                             unsigned short* __restrict__ o_wih_f, unsigned short* __restrict__ o_whh_f,
                             unsigned short* __restrict__ o_wih_b, unsigned short* __restrict__ o_whh_b,
                             unsigned short* __restrict__ o_wconv) {
    int i = blockIdx.x * 256 + threadIdx.x;
    if (i < G_ * H_) {
        o_wih_f[i] = f2bf(wih_f[i]);
        o_whh_f[i] = f2bf(whh_f[i]);
        o_wih_b[i] = f2bf(wih_b[i]);
        o_whh_b[i] = f2bf(whh_b[i]);
    }
    if (i < 3 * 2 * 32 * 16) {   // 3072
        int j    = i & 15;
        int lane = (i >> 4) & 31;
        int w2   = (i >> 9) & 1;
        int kk   = i >> 10;
        int ch = w2 * 16 + (lane & 15);
        int e  = (lane >> 4) * 16 + j;
        o_wconv[i] = f2bf(conv_w[(ch * E_ + e) * 3 + kk]);
    }
}

// ---------------------------------------------------------------------------
// K1: embed + conv1d(k=3,pad=1) + bias + maxpool(2) -> seqT [T][B][C] bf16
// grid: (L/32, B), block 128 (4 waves). Wave w: postile=w>>1, chtile=w&1.
// Conv done as 3 accumulated WMMAs over K=E=32; weights pre-swizzled.
// ---------------------------------------------------------------------------
__global__ void __launch_bounds__(128) prep_kernel(const int* __restrict__ x,
                                                   const float* __restrict__ emb,
                                                   const unsigned short* __restrict__ fragW,
                                                   const float* __restrict__ conv_b,
                                                   unsigned short* __restrict__ seqT) {
    __shared__ __align__(16) unsigned short ebuf[34][32];   // emb rows, bf16

    int tid = threadIdx.x, wave = tid >> 5, lid = tid & 31;
    int b = blockIdx.y;
    int lbase = blockIdx.x * 32;

    // stage embedding rows for positions lbase-1 .. lbase+32 (zero pad OOB)
    // 34 rows * 8 quads, vectorized float4 -> ushort4
    for (int i = tid; i < 34 * 8; i += 128) {
        int srow = i >> 3, eq = i & 7;
        int p = lbase - 1 + srow;
        u16x4 o = {0, 0, 0, 0};
        if (p >= 0 && p < L_) {
            int idx = x[b * L_ + p];
            f32x4 v = *(const f32x4*)(emb + idx * E_ + eq * 4);
            o[0] = f2bf(v[0]); o[1] = f2bf(v[1]);
            o[2] = f2bf(v[2]); o[3] = f2bf(v[3]);
        }
        *(u16x4*)&ebuf[srow][eq * 4] = o;
    }
    __syncthreads();

    int postile = wave >> 1, chtile = wave & 1;
    int ch = chtile * 16 + (lid & 15);
    float cb = conv_b[ch];

    f32x8 acc;
#pragma unroll
    for (int v = 0; v < 8; ++v) acc[v] = cb;

#pragma unroll
    for (int kk = 0; kk < 3; ++kk) {
        // A[m][e] = emb row for position (lbase + postile*16 + m) - 1 + kk
        bhalf16 a = load_a_frag(&ebuf[postile * 16 + kk][0], 32, lid);
        bhalf16 bf = load_b_frag_swz(fragW + (kk * 2 + chtile) * 32 * 16, lid);
        acc = wmma_bf16(a, bf, acc);
    }

    // maxpool over position pairs (m = v + 8*hi), store seqT[t][b][ch]
    int hi = lid >> 4;
#pragma unroll
    for (int j = 0; j < 4; ++j) {
        float pm = fmaxf(acc[2 * j], acc[2 * j + 1]);
        int u = hi * 4 + j;
        int t = blockIdx.x * 16 + postile * 8 + u;
        seqT[((size_t)t * B_ + b) * C_ + ch] = f2bf(pm);
    }
}

// ---------------------------------------------------------------------------
// K2: LSTM. grid (B/16, 2), block 256 (8 waves = 8 gate tiles of 4H=128).
// dir 0: full forward scan over T=256, write final h.  dir 1: single step at
// t=T-1 from zero state (matches reference hs_b[0]).
// ---------------------------------------------------------------------------
__global__ void __launch_bounds__(256) lstm_kernel(const unsigned short* __restrict__ seqT,
                                                   const unsigned short* __restrict__ wih_f,
                                                   const unsigned short* __restrict__ whh_f,
                                                   const float* __restrict__ bihf, const float* __restrict__ bhhf,
                                                   const unsigned short* __restrict__ wih_b,
                                                   const unsigned short* __restrict__ whh_b,
                                                   const float* __restrict__ bihb, const float* __restrict__ bhhb,
                                                   float* __restrict__ hf_out, float* __restrict__ hb_out) {
    __shared__ __align__(16) float gates[16][G_ + 4];
    __shared__ __align__(16) unsigned short hbuf[16][H_];  // bf16 hidden (A-matrix src)
    __shared__ __align__(16) float cbuf[16][H_];

    int tid = threadIdx.x, wave = tid >> 5, lid = tid & 31;
    int dir = blockIdx.y;
    int bb = blockIdx.x * 16;

    const unsigned short* wih = dir ? wih_b : wih_f;
    const unsigned short* whh = dir ? whh_b : whh_f;
    const float* bih = dir ? bihb : bihf;
    const float* bhh = dir ? bhhb : bhhf;
    float* hout = dir ? hb_out : hf_out;

    int n_g = wave * 16 + (lid & 15);
    float bias = bih[n_g] + bhh[n_g];

    bhalf16 bw_ih = load_b_frag_wT(wih + (wave * 16) * 32, lid);
    bhalf16 bw_hh = load_b_frag_wT(whh + (wave * 16) * 32, lid);

    // zero initial state (512 elements over 256 threads)
    {
        int e = tid;
        hbuf[e >> 5][e & 31] = 0; cbuf[e >> 5][e & 31] = 0.0f;
        e += 256;
        hbuf[e >> 5][e & 31] = 0; cbuf[e >> 5][e & 31] = 0.0f;
    }
    __syncthreads();

    int t0 = dir ? (T_ - 1) : 0;
    int nsteps = dir ? 1 : T_;

    for (int s = 0; s < nsteps; ++s) {
        int t = t0 + s;
        const unsigned short* abase = seqT + ((size_t)t * B_ + bb) * C_;
        bhalf16 ax = load_a_frag(abase, 32, lid);

        // prefetch next step's seq tile (global_prefetch_b8)
        if (s + 1 < nsteps)
            __builtin_prefetch(seqT + ((size_t)(t + 1) * B_ + bb) * C_ + lid * 2, 0, 1);

        f32x8 acc;
#pragma unroll
        for (int v = 0; v < 8; ++v) acc[v] = bias;
        acc = wmma_bf16(ax, bw_ih, acc);

        bhalf16 ah = load_a_frag(&hbuf[0][0], 32, lid);
        acc = wmma_bf16(ah, bw_hh, acc);

        int hi = lid >> 4, nn = wave * 16 + (lid & 15);
#pragma unroll
        for (int v = 0; v < 8; ++v) gates[v + 8 * hi][nn] = acc[v];
        __syncthreads();

        bool last = (s == nsteps - 1);
#pragma unroll
        for (int r = 0; r < 2; ++r) {
            int e = tid + r * 256;
            int m = e >> 5, n = e & 31;
            float gi = gates[m][n];
            float gf = gates[m][32 + n];
            float gg = gates[m][64 + n];
            float go = gates[m][96 + n];
            float c = sigm(gf) * cbuf[m][n] + sigm(gi) * tanh_(gg);
            float h = sigm(go) * tanh_(c);
            cbuf[m][n] = c;
            hbuf[m][n] = f2bf(h);
            if (last) hout[(size_t)(bb + m) * H_ + n] = h;
        }
        __syncthreads();
    }
}

// ---------------------------------------------------------------------------
// K3: out[b] = [h_f[b], h_b[b]] @ fc_w^T + fc_b    (f32, tiny)
// ---------------------------------------------------------------------------
__global__ void fc_kernel(const float* __restrict__ hf, const float* __restrict__ hb,
                          const float* __restrict__ fcw, const float* __restrict__ fcb,
                          float* __restrict__ out) {
    int b = blockIdx.x * 256 + threadIdx.x;
    if (b >= B_) return;
    float a0 = fcb[0], a1 = fcb[1], a2 = fcb[2];
#pragma unroll 8
    for (int k = 0; k < 32; ++k) {
        float v = hf[b * 32 + k];
        a0 += v * fcw[0 * 64 + k];
        a1 += v * fcw[1 * 64 + k];
        a2 += v * fcw[2 * 64 + k];
    }
#pragma unroll 8
    for (int k = 0; k < 32; ++k) {
        float v = hb[b * 32 + k];
        a0 += v * fcw[0 * 64 + 32 + k];
        a1 += v * fcw[1 * 64 + 32 + k];
        a2 += v * fcw[2 * 64 + 32 + k];
    }
    out[b * 3 + 0] = a0;
    out[b * 3 + 1] = a1;
    out[b * 3 + 2] = a2;
}

extern "C" void kernel_launch(void* const* d_in, const int* in_sizes, int n_in,
                              void* d_out, int out_size, void* d_ws, size_t ws_size,
                              hipStream_t stream) {
    const int*   x      = (const int*)d_in[0];
    const float* emb    = (const float*)d_in[1];
    const float* conv_w = (const float*)d_in[2];
    const float* conv_b = (const float*)d_in[3];
    const float* w_ih_f = (const float*)d_in[4];
    const float* w_hh_f = (const float*)d_in[5];
    const float* b_ih_f = (const float*)d_in[6];
    const float* b_hh_f = (const float*)d_in[7];
    const float* w_ih_b = (const float*)d_in[8];
    const float* w_hh_b = (const float*)d_in[9];
    const float* b_ih_b = (const float*)d_in[10];
    const float* b_hh_b = (const float*)d_in[11];
    const float* fc_w   = (const float*)d_in[12];
    const float* fc_b   = (const float*)d_in[13];
    float* out = (float*)d_out;

    // workspace carve-out
    auto al = [](size_t v) { return (v + 255) & ~(size_t)255; };
    unsigned char* ws = (unsigned char*)d_ws;
    size_t off = 0;
    unsigned short* bw_ih_f = (unsigned short*)(ws + off); off += al(G_ * H_ * 2);
    unsigned short* bw_hh_f = (unsigned short*)(ws + off); off += al(G_ * H_ * 2);
    unsigned short* bw_ih_b = (unsigned short*)(ws + off); off += al(G_ * H_ * 2);
    unsigned short* bw_hh_b = (unsigned short*)(ws + off); off += al(G_ * H_ * 2);
    unsigned short* bwconv  = (unsigned short*)(ws + off); off += al(3 * E_ * C_ * 2);
    unsigned short* seqT    = (unsigned short*)(ws + off); off += al((size_t)T_ * B_ * C_ * 2);
    float* h_f = (float*)(ws + off); off += al((size_t)B_ * H_ * 4);
    float* h_b = (float*)(ws + off); off += al((size_t)B_ * H_ * 4);

    wprep_kernel<<<16, 256, 0, stream>>>(w_ih_f, w_hh_f, w_ih_b, w_hh_b, conv_w,
                                         bw_ih_f, bw_hh_f, bw_ih_b, bw_hh_b, bwconv);

    prep_kernel<<<dim3(L_ / 32, B_), 128, 0, stream>>>(x, emb, bwconv, conv_b, seqT);

    lstm_kernel<<<dim3(B_ / 16, 2), 256, 0, stream>>>(seqT,
                                                      bw_ih_f, bw_hh_f, b_ih_f, b_hh_f,
                                                      bw_ih_b, bw_hh_b, b_ih_b, b_hh_b,
                                                      h_f, h_b);

    fc_kernel<<<B_ / 256, 256, 0, stream>>>(h_f, h_b, fc_w, fc_b, out);
}
